// TernaryLinear_82325933130246
// MI455X (gfx1250) — matrix operand order repaired and checked by reading, not compiled, not run on hardware
//
#include <hip/hip_runtime.h>
#include <hip/hip_bf16.h>

// Ternary linear:  out[M,N] = x[M,K] @ q(w[N,K])^T,  q(w) in {-1,0,+1}
// M = 8192, N = 4096, K = 4096, fp32 in/out.
//
// Fast path (needs ws >= (M+N)*K*2 bytes):
//   1) quantize w -> ternary f16 in ws   (bandwidth-bound, ~4 us)
//   2) convert  x -> f16 in ws           (bandwidth-bound, ~9 us)
//   3) f16 WMMA GEMM, global->LDS via GLOBAL_LOAD_ASYNC_TO_LDS_B128 (ASYNCcnt),
//      v_wmma_f32_16x16x32_f16 with fp32 accumulation (compute-bound, ~50+ us)
// Fallback: fused single-kernel version (round-0, on-the-fly conversion).

typedef _Float16 v16h __attribute__((ext_vector_type(16)));
typedef _Float16 v8h  __attribute__((ext_vector_type(8)));
typedef _Float16 v4h  __attribute__((ext_vector_type(4)));
typedef float    v8f  __attribute__((ext_vector_type(8)));
typedef float    v4f  __attribute__((ext_vector_type(4)));

#define BM 128
#define BN 128

// ---------------------------------------------------------------------------
// Prep kernels: fp32 -> f16 (x) and fp32 -> ternary f16 (w)
// ---------------------------------------------------------------------------
__global__ __launch_bounds__(256) void convert_x_f16(
    const float* __restrict__ X, _Float16* __restrict__ Xh) {
  const int i = blockIdx.x * 256 + threadIdx.x;   // float4 index
  const v4f v = ((const v4f*)X)[i];
  v4h h;
#pragma unroll
  for (int j = 0; j < 4; ++j) h[j] = (_Float16)v[j];
  ((v4h*)Xh)[i] = h;
}

__global__ __launch_bounds__(256) void quantize_w_f16(
    const float* __restrict__ W, _Float16* __restrict__ Wh) {
  const int i = blockIdx.x * 256 + threadIdx.x;   // float4 index
  const v4f v = ((const v4f*)W)[i];
  v4h h;
#pragma unroll
  for (int j = 0; j < 4; ++j) {
    const float w = v[j];
    h[j] = (_Float16)((w > 0.5f) ? 1.0f : ((w < -0.5f) ? -1.0f : 0.0f));
  }
  ((v4h*)Wh)[i] = h;
}

// ---------------------------------------------------------------------------
// Async copy helpers (CDNA5): 16B global -> LDS per lane, tracked by ASYNCcnt
// ---------------------------------------------------------------------------
__device__ __forceinline__ void async_ld_b128(unsigned lds_off, unsigned goff,
                                              const void* sbase) {
  asm volatile("global_load_async_to_lds_b128 %0, %1, %2"
               :: "v"(lds_off), "v"(goff), "s"(sbase) : "memory");
}
__device__ __forceinline__ void wait_async0() {
  asm volatile("s_wait_asynccnt 0x0" ::: "memory");
}
// low 32 bits of a generic shared pointer == LDS byte offset (aperture rule)
__device__ __forceinline__ unsigned lds_off32(const void* p) {
  return (unsigned)(unsigned long long)p;
}

// ---------------------------------------------------------------------------
// Fast GEMM: f16 operands, async LDS staging, double-buffered, BK=64
// ---------------------------------------------------------------------------
#define FBK 64
#define FLDK 72  // padded row stride in halves: 144 B (16B-aligned)

__global__ __launch_bounds__(256)
void ternary_wmma_gemm_async(const _Float16* __restrict__ Xh,  // [M,K]
                             const _Float16* __restrict__ Wh,  // [N,K] ternary
                             float* __restrict__ Out,          // [M,N]
                             int M, int N, int K) {
  __shared__ __attribute__((aligned(16))) _Float16 sA[2][BM * FLDK];
  __shared__ __attribute__((aligned(16))) _Float16 sB[2][BN * FLDK];

  const int tid  = threadIdx.x;
  const int lane = tid & 31;
  const int wave = tid >> 5;
  const int wm   = wave & 3;        // 4 waves along M -> 32 rows each
  const int wn   = wave >> 2;       // 2 waves along N -> 64 cols each
  const int bm   = blockIdx.y * BM;
  const int bn   = blockIdx.x * BN;

  const _Float16* gA = Xh + (size_t)bm * K;
  const _Float16* gB = Wh + (size_t)bn * K;
  const unsigned sA0 = lds_off32(&sA[0][0]);
  const unsigned sB0 = lds_off32(&sB[0][0]);
  const unsigned stgA = (unsigned)(BM * FLDK * 2);
  const unsigned stgB = (unsigned)(BN * FLDK * 2);

  // Stage copy: tile = 128 rows x 64 halves = 128 rows x 8 x16B chunks.
  // 1024 chunks per tile, 256 threads -> 4 chunks each, for A and B.
  auto aload = [&](int kt, int s) {
    const unsigned kb = (unsigned)(kt * FBK) * 2u;   // byte offset along K
#pragma unroll
    for (int i = 0; i < 4; ++i) {
      const int c   = tid + i * 256;
      const int row = c >> 3;
      const int ch  = c & 7;
      const unsigned go = (unsigned)row * (unsigned)(K * 2) + kb + ch * 16;
      const unsigned lo = (unsigned)row * (FLDK * 2) + ch * 16;
      async_ld_b128(sA0 + s * stgA + lo, go, gA);
      async_ld_b128(sB0 + s * stgB + lo, go, gB);
    }
  };

  // WMMA fragment addressing (contiguous k order: lane half h holds k 16h..16h+15 for A and B)
  const int frow  = lane & 15;
  const int koffA = (lane >> 4) * 16;   // A 16x32: K {0..15} / {16..31}
  const int koffB = (lane >> 4) * 16;   // B 32x16: K {0..15} / {16..31}, col = lane&15

  v8f acc[2][4] = {};

  auto compute = [&](int s) {
#pragma unroll
    for (int ks = 0; ks < 2; ++ks) {    // two 32-wide K steps per stage
      const int kb = ks * 32;
      v16h af[2], bf[4];
#pragma unroll
      for (int i = 0; i < 2; ++i) {
        const _Float16* p = &sA[s][(wm * 32 + i * 16 + frow) * FLDK + kb + koffA];
        v8h lo = *(const v8h*)p;
        v8h hi = *(const v8h*)(p + 8);
        af[i] = __builtin_shufflevector(lo, hi,
                    0,1,2,3,4,5,6,7,8,9,10,11,12,13,14,15);
      }
#pragma unroll
      for (int j = 0; j < 4; ++j) {
        const _Float16* p = &sB[s][(wn * 64 + j * 16 + frow) * FLDK + kb + koffB];
        bf[j] = *(const v16h*)p;        // 32B contiguous -> 2x ds_load_b128
      }
#pragma unroll
      for (int i = 0; i < 2; ++i)
#pragma unroll
        for (int j = 0; j < 4; ++j)
          acc[i][j] = __builtin_amdgcn_wmma_f32_16x16x32_f16(
              false, af[i], false, bf[j], (short)0, acc[i][j], false, false);
    }
  };

  const int nk = K / FBK;
  aload(0, 0);
  wait_async0();
  __syncthreads();
  for (int kt = 0; kt < nk; ++kt) {
    const int cur = kt & 1;
    if (kt + 1 < nk) aload(kt + 1, cur ^ 1);  // async engine fills next buffer
    compute(cur);                             // WMMA on current buffer
    if (kt + 1 < nk) {
      wait_async0();                          // own wave's copies landed
      __syncthreads();                        // everyone's copies visible
    }
  }

  // Epilogue: C/D 16x16 f32 layout (VGPR r -> row r / r+8)
  const int rsel = (lane >> 4) * 8;
  const int crow = bm + wm * 32 + rsel;
  const int ccol = bn + wn * 64 + (lane & 15);
#pragma unroll
  for (int i = 0; i < 2; ++i)
#pragma unroll
    for (int j = 0; j < 4; ++j) {
      float* o = Out + (size_t)(crow + i * 16) * N + (ccol + j * 16);
#pragma unroll
      for (int r = 0; r < 8; ++r)
        o[(size_t)r * N] = acc[i][j][r];
    }
}

// ---------------------------------------------------------------------------
// Fallback: fused single-kernel (round-0 proven), BK=32, on-the-fly convert
// ---------------------------------------------------------------------------
#define SBK 32
#define SLDK 56

__global__ __launch_bounds__(256)
void ternary_wmma_gemm_fused(const float* __restrict__ X,
                             const float* __restrict__ W,
                             float* __restrict__ Out,
                             int M, int N, int K) {
  __shared__ __attribute__((aligned(16))) _Float16 sA[2][BM * SLDK];
  __shared__ __attribute__((aligned(16))) _Float16 sB[2][BN * SLDK];

  const int tid  = threadIdx.x;
  const int lane = tid & 31;
  const int wave = tid >> 5;
  const int wm   = wave & 3;
  const int wn   = wave >> 2;
  const int bm   = blockIdx.y * BM;
  const int bn   = blockIdx.x * BN;

  const int Kv = K >> 2;
  const v4f* Xg = (const v4f*)(X + (size_t)bm * K);
  const v4f* Wg = (const v4f*)(W + (size_t)bn * K);

  v4f ra[4], rb[4];
  auto gload = [&](int kt) {
    const int k4 = (kt * SBK) >> 2;
#pragma unroll
    for (int i = 0; i < 4; ++i) {
      const int f = tid + i * 256;
      const int row = f >> 3;
      const int c4  = f & 7;
      ra[i] = Xg[(size_t)row * Kv + k4 + c4];
      rb[i] = Wg[(size_t)row * Kv + k4 + c4];
    }
  };
  auto lstore = [&](int s) {
#pragma unroll
    for (int i = 0; i < 4; ++i) {
      const int f = tid + i * 256;
      const int row = f >> 3;
      const int c4  = f & 7;
      v4h ha, hb;
#pragma unroll
      for (int j = 0; j < 4; ++j) {
        ha[j] = (_Float16)ra[i][j];
        const float w = rb[i][j];
        hb[j] = (_Float16)((w > 0.5f) ? 1.0f : ((w < -0.5f) ? -1.0f : 0.0f));
      }
      *(v4h*)&sA[s][row * SLDK + c4 * 4] = ha;
      *(v4h*)&sB[s][row * SLDK + c4 * 4] = hb;
    }
  };

  const int frow  = lane & 15;
  const int koffA = (lane >> 4) * 16;
  const int koffB = (lane >> 4) * 16;

  v8f acc[2][4] = {};
  auto compute = [&](int s) {
    v16h af[2], bf[4];
#pragma unroll
    for (int i = 0; i < 2; ++i) {
      const _Float16* p = &sA[s][(wm * 32 + i * 16 + frow) * SLDK + koffA];
      v8h lo = *(const v8h*)p;
      v8h hi = *(const v8h*)(p + 8);
      af[i] = __builtin_shufflevector(lo, hi,
                  0,1,2,3,4,5,6,7,8,9,10,11,12,13,14,15);
    }
#pragma unroll
    for (int j = 0; j < 4; ++j) {
      const _Float16* p = &sB[s][(wn * 64 + j * 16 + frow) * SLDK + koffB];
      bf[j] = *(const v16h*)p;
    }
#pragma unroll
    for (int i = 0; i < 2; ++i)
#pragma unroll
      for (int j = 0; j < 4; ++j)
        acc[i][j] = __builtin_amdgcn_wmma_f32_16x16x32_f16(
            false, af[i], false, bf[j], (short)0, acc[i][j], false, false);
  };

  const int nk = K / SBK;
  gload(0);
  lstore(0);
  __syncthreads();
  for (int kt = 0; kt < nk; ++kt) {
    const int cur = kt & 1;
    if (kt + 1 < nk) gload(kt + 1);
    compute(cur);
    if (kt + 1 < nk) {
      lstore(cur ^ 1);
      __syncthreads();
    }
  }

  const int rsel = (lane >> 4) * 8;
  const int crow = bm + wm * 32 + rsel;
  const int ccol = bn + wn * 64 + (lane & 15);
#pragma unroll
  for (int i = 0; i < 2; ++i)
#pragma unroll
    for (int j = 0; j < 4; ++j) {
      float* o = Out + (size_t)(crow + i * 16) * N + (ccol + j * 16);
#pragma unroll
      for (int r = 0; r < 8; ++r)
        o[(size_t)r * N] = acc[i][j][r];
    }
}

// ---------------------------------------------------------------------------
extern "C" void kernel_launch(void* const* d_in, const int* in_sizes, int n_in,
                              void* d_out, int out_size, void* d_ws, size_t ws_size,
                              hipStream_t stream) {
  const float* x = (const float*)d_in[0];   // [B,S,K] fp32
  const float* w = (const float*)d_in[1];   // [N,K]   fp32
  float* out = (float*)d_out;               // [B,S,N] fp32

  const int K = 4096;
  const int M = in_sizes[0] / K;            // 8192
  const int N = in_sizes[1] / K;            // 4096

  const size_t xBytes = (size_t)M * K * sizeof(_Float16);
  const size_t wBytes = (size_t)N * K * sizeof(_Float16);
  dim3 grid(N / BN, M / BM);

  if (ws_size >= xBytes + wBytes) {
    _Float16* xh = (_Float16*)d_ws;
    _Float16* wh = (_Float16*)((char*)d_ws + xBytes);
    convert_x_f16 <<<(M / 4) * (K / 256), 256, 0, stream>>>(x, xh);
    quantize_w_f16<<<(N / 4) * (K / 256), 256, 0, stream>>>(w, wh);
    ternary_wmma_gemm_async<<<grid, 256, 0, stream>>>(xh, wh, out, M, N, K);
  } else {
    ternary_wmma_gemm_fused<<<grid, 256, 0, stream>>>(x, w, out, M, N, K);
  }
}
